// PointNet_main_69020124446965
// MI455X (gfx1250) — compile-verified, matrix operand order checked by reading, not run on hardware
//
#include <hip/hip_runtime.h>
#include <hip/hip_bf16.h>

#define B_ 16
#define N_ 2048
#define E_ 1024
#define CMAX_ 128
#define BN_EPS_ 1e-5f

typedef __attribute__((ext_vector_type(16))) _Float16 v16h;
typedef __attribute__((ext_vector_type(8)))  float    v8f;

union V16HU { v16h v; uint4 q[2]; };
union V8FU  { v8f  v; float f[8]; };

#define WMMA16(a, b, c) __builtin_amdgcn_wmma_f32_16x16x32_f16(false, (a), false, (b), (short)0, (c), false, false)

// Load a 16x32 f16 fragment (A-layout per CDNA5 ISA; B mirrors with N as "row").
// Storage: row-major, K contiguous, ldk multiple of 8, base 16B aligned.
__device__ inline v16h ld_frag(const _Float16* __restrict__ base, int row0, int ldk, int k) {
  int lane = threadIdx.x & 31;
  int r = lane & 15;
  int hi = lane >> 4;
  const _Float16* p = base + (size_t)(row0 + r) * (size_t)ldk + (size_t)(k + hi * 8);
  V16HU u;
  u.q[0] = *(const uint4*)(p);
  u.q[1] = *(const uint4*)(p + 16);
  return u.v;
}

// ---------------- init / density ----------------

__global__ void k_init(unsigned* gmax, unsigned* h5max) {
  int i = blockIdx.x * 256 + threadIdx.x;
  if (i < B_ * E_) h5max[i] = 0u;
  if (i == 0) *gmax = 0u;
}

__global__ void k_norms(const float* __restrict__ x, float* __restrict__ nrm) {
  int i = blockIdx.x * 256 + threadIdx.x;   // over B*N
  int n = i % N_;
  int b = i / N_;
  const float* xb = x + (size_t)b * 3 * N_;
  float a = xb[n], c = xb[N_ + n], d = xb[2 * N_ + n];
  nrm[i] = sqrtf(a * a + c * c + d * d);
}

// One block per (b,n): 2048 distances in LDS, radix-select 32nd smallest,
// exact mean of the 32 smallest, inv density + global atomic max.
__global__ void k_density(const float* __restrict__ x, const float* __restrict__ nrm,
                          float* __restrict__ dens, unsigned* __restrict__ gmax) {
  __shared__ float dist[N_];
  __shared__ unsigned hist[256];
  __shared__ float redf[256];
  __shared__ unsigned redu[256];
  __shared__ unsigned s_sel, s_rem;
  int b = blockIdx.x / N_;
  int n = blockIdx.x % N_;
  int tid = threadIdx.x;
  const float* xb = x + (size_t)b * 3 * N_;
  float px = xb[n], py = xb[N_ + n], pz = xb[2 * N_ + n];
  float nn = nrm[b * N_ + n];
  for (int m = tid; m < N_; m += 256) {
    float inner = px * xb[m] + py * xb[N_ + m] + pz * xb[2 * N_ + m];
    dist[m] = 1.0f - inner / (nn * nrm[b * N_ + m]);
  }
  __syncthreads();
  unsigned prefix = 0, rem = 32;
  for (int p = 0; p < 4; ++p) {
    int shift = 24 - 8 * p;
    hist[tid] = 0;
    __syncthreads();
    unsigned himask = (p == 0) ? 0u : (0xFFFFFFFFu << (shift + 8));
    for (int m = tid; m < N_; m += 256) {
      unsigned u = __float_as_uint(dist[m]);
      u = (u & 0x80000000u) ? ~u : (u | 0x80000000u);
      if ((u & himask) == (prefix & himask))
        atomicAdd(&hist[(u >> shift) & 255u], 1u);
    }
    __syncthreads();
    if (tid == 0) {
      unsigned cum = 0, sel = 255, r2 = rem;
      for (int bin = 0; bin < 256; ++bin) {
        if (cum + hist[bin] >= rem) { sel = (unsigned)bin; r2 = rem - cum; break; }
        cum += hist[bin];
      }
      s_sel = sel; s_rem = r2;
    }
    __syncthreads();
    prefix |= (s_sel << shift);
    rem = s_rem;
    __syncthreads();
  }
  float partial = 0.f; unsigned cnt = 0;
  for (int m = tid; m < N_; m += 256) {
    unsigned u = __float_as_uint(dist[m]);
    u = (u & 0x80000000u) ? ~u : (u | 0x80000000u);
    if (u < prefix) { partial += dist[m]; cnt++; }
  }
  redf[tid] = partial; redu[tid] = cnt;
  __syncthreads();
  for (int s = 128; s > 0; s >>= 1) {
    if (tid < s) { redf[tid] += redf[tid + s]; redu[tid] += redu[tid + s]; }
    __syncthreads();
  }
  if (tid == 0) {
    unsigned u = prefix;
    unsigned fb = (u & 0x80000000u) ? (u & 0x7FFFFFFFu) : ~u;
    float T = __uint_as_float(fb);
    float total = redf[0] + T * (float)(32 - (int)redu[0]);
    float invd = 32.0f / total;            // 1 / mean(32 smallest)
    dens[b * N_ + n] = invd;
    atomicMax(gmax, __float_as_uint(invd)); // positive floats order as uints
  }
}

__global__ void k_normalize(float* dens, const unsigned* gmax) {
  int i = blockIdx.x * 256 + threadIdx.x;
  dens[i] /= __uint_as_float(*gmax);
}

__global__ void k_batchmax(const float* __restrict__ dens, float* __restrict__ dmaxb) {
  __shared__ float red[256];
  int b = blockIdx.x, tid = threadIdx.x;
  float mx = 0.f;
  for (int n = tid; n < N_; n += 256) mx = fmaxf(mx, dens[b * N_ + n]);
  red[tid] = mx; __syncthreads();
  for (int s = 128; s > 0; s >>= 1) {
    if (tid < s) red[tid] = fmaxf(red[tid], red[tid + s]);
    __syncthreads();
  }
  if (tid == 0) dmaxb[b] = red[0];
}

// ---------------- attention (rank-1 energy) ----------------

__global__ void k_sumsq(const float* wqk, int c4, float* S) {
  if (threadIdx.x == 0) {
    float s = 0.f;
    for (int i = 0; i < c4; ++i) s += wqk[i] * wqk[i];
    *S = s;
  }
}

// rowZ[b,n] = sum_m exp(S*d_n*(d_m - dmax_b))
__global__ void k_rowZ(const float* __restrict__ dens, const float* __restrict__ dmaxb,
                       const float* __restrict__ Sp, float* __restrict__ rowZ) {
  __shared__ float sd[N_];
  int i = blockIdx.x * 256 + threadIdx.x;
  int b = i / N_;
  for (int m = threadIdx.x; m < N_; m += 256) sd[m] = dens[b * N_ + m];
  __syncthreads();
  float a = Sp[0] * dens[i];
  float dm = dmaxb[b];
  float z = 0.f;
  for (int m = 0; m < N_; ++m) z += __expf(a * (sd[m] - dm));
  rowZ[i] = z;
}

// w16[b][m][n] = exp(S*d_n*(d_m - dmax_b)) / rowZ[n]  (= attn[n,m], f16, L2-resident)
// colsum[b][m] = sum_n attn[n,m]
__global__ void k_build_w(const float* __restrict__ dens, const float* __restrict__ rowZ,
                          const float* __restrict__ dmaxb, const float* __restrict__ Sp,
                          _Float16* __restrict__ w16, float* __restrict__ colsum) {
  __shared__ float sd[N_];
  __shared__ float sz[N_];
  int blocksPerB = N_ / 8;
  int b = blockIdx.x / blocksPerB;
  int m0 = (blockIdx.x % blocksPerB) * 8;
  int tid = threadIdx.x;
  for (int n = tid; n < N_; n += 256) { sd[n] = dens[b * N_ + n]; sz[n] = 1.0f / rowZ[b * N_ + n]; }
  __syncthreads();
  int tx = tid & 31, ty = tid >> 5;
  int m = m0 + ty;
  float cm = Sp[0] * (dens[b * N_ + m] - dmaxb[b]);
  _Float16* wrow = w16 + ((size_t)b * N_ + m) * N_;
  float cs = 0.f;
  for (int n0 = 0; n0 < N_; n0 += 32) {
    int n = n0 + tx;
    float e = __expf(cm * sd[n]) * sz[n];
    wrow[n] = (_Float16)e;
    cs += e;
  }
  for (int off = 16; off; off >>= 1) cs += __shfl_xor(cs, off);
  if (tx == 0) colsum[b * N_ + m] = cs;
}

// x_r[b,c,m] = (sum_n x16[c,n] * attn[n,m]) / (1e-9 + colsum[m])
// 2x4 register-blocked WMMA: 6 fragment loads feed 8 wmma per k-step.
__global__ void k_xr(const _Float16* __restrict__ h16cn, const _Float16* __restrict__ w16,
                     const float* __restrict__ colsum, float* __restrict__ xr, int C) {
  int b = blockIdx.z;
  int c0 = blockIdx.y * 32;
  int m0 = blockIdx.x * 64;
  const _Float16* A = h16cn + (size_t)b * C * N_;
  const _Float16* Bm = w16 + (size_t)b * N_ * N_;
  V8FU acc[2][4];
#pragma unroll
  for (int i = 0; i < 2; ++i)
#pragma unroll
    for (int j = 0; j < 4; ++j) acc[i][j].v = (v8f){};
  for (int k = 0; k < N_; k += 32) {
    v16h a0 = ld_frag(A, c0,      N_, k);
    v16h a1 = ld_frag(A, c0 + 16, N_, k);
    v16h b0 = ld_frag(Bm, m0,      N_, k);
    v16h b1 = ld_frag(Bm, m0 + 16, N_, k);
    v16h b2 = ld_frag(Bm, m0 + 32, N_, k);
    v16h b3 = ld_frag(Bm, m0 + 48, N_, k);
    acc[0][0].v = WMMA16(a0, b0, acc[0][0].v);
    acc[0][1].v = WMMA16(a0, b1, acc[0][1].v);
    acc[0][2].v = WMMA16(a0, b2, acc[0][2].v);
    acc[0][3].v = WMMA16(a0, b3, acc[0][3].v);
    acc[1][0].v = WMMA16(a1, b0, acc[1][0].v);
    acc[1][1].v = WMMA16(a1, b1, acc[1][1].v);
    acc[1][2].v = WMMA16(a1, b2, acc[1][2].v);
    acc[1][3].v = WMMA16(a1, b3, acc[1][3].v);
  }
  int lane = threadIdx.x;
  int ml = lane & 15;
  int chalf = (lane >> 4) * 8;
#pragma unroll
  for (int j = 0; j < 4; ++j) {
    int m = m0 + j * 16 + ml;
    float inv = 1.0f / (colsum[b * N_ + m] + 1e-9f);
#pragma unroll
    for (int i = 0; i < 2; ++i)
#pragma unroll
      for (int r = 0; r < 8; ++r) {
        int c = c0 + i * 16 + r + chalf;
        xr[((size_t)b * C + c) * N_ + m] = acc[i][j].f[r] * inv;
      }
  }
}

// d16[b][n][c] = f16(h - x_r)
__global__ void k_diff16(const float* __restrict__ h, const float* __restrict__ xr,
                         _Float16* __restrict__ d16, int C) {
  size_t i = (size_t)blockIdx.x * 256 + threadIdx.x;
  size_t total = (size_t)B_ * C * N_;
  if (i >= total) return;
  int n = (int)(i % N_);
  int c = (int)((i / N_) % C);
  size_t b = i / ((size_t)C * N_);
  d16[(b * N_ + n) * C + c] = (_Float16)(h[i] - xr[i]);
}

// hout = h + relu(bn(wt @ (h - x_r) + bt))  -- 1x4 blocked WMMA (A-frag reuse)
__global__ void k_post(const _Float16* __restrict__ wt16, const _Float16* __restrict__ d16,
                       const float* __restrict__ h, const float* __restrict__ bt,
                       const float* __restrict__ bnp, float* __restrict__ hout, int C) {
  int b = blockIdx.z, c0 = blockIdx.y * 16, p0 = blockIdx.x * 64;
  const _Float16* Bm = d16 + (size_t)b * N_ * C;
  V8FU acc[4];
#pragma unroll
  for (int j = 0; j < 4; ++j) acc[j].v = (v8f){};
  for (int k = 0; k < C; k += 32) {
    v16h a = ld_frag(wt16, c0, C, k);
#pragma unroll
    for (int j = 0; j < 4; ++j) {
      v16h bf = ld_frag(Bm, p0 + j * 16, C, k);
      acc[j].v = WMMA16(a, bf, acc[j].v);
    }
  }
  int lane = threadIdx.x;
  int pl = lane & 15;
  int chalf = (lane >> 4) * 8;
#pragma unroll
  for (int j = 0; j < 4; ++j) {
    int p = p0 + j * 16 + pl;
#pragma unroll
    for (int r = 0; r < 8; ++r) {
      int c = c0 + r + chalf;
      float y = acc[j].f[r] + bt[c];
      float g = bnp[c], bb = bnp[C + c], mm = bnp[2 * C + c], vv = bnp[3 * C + c];
      float t = fmaxf(g * (y - mm) * rsqrtf(vv + BN_EPS_) + bb, 0.f);
      size_t idx = ((size_t)b * C + c) * N_ + p;
      hout[idx] = h[idx] + t;
    }
  }
}

// ---------------- convs ----------------

__global__ void k_conv1(const float* __restrict__ x, const float* __restrict__ w1,
                        const float* __restrict__ bn1, float* __restrict__ h) {
  int i = blockIdx.x * 256 + threadIdx.x;   // over B*64*N
  int n = i % N_;
  int c = (i / N_) & 63;
  int b = i / (64 * N_);
  const float* xb = x + (size_t)b * 3 * N_;
  float v = w1[c * 3] * xb[n] + w1[c * 3 + 1] * xb[N_ + n] + w1[c * 3 + 2] * xb[2 * N_ + n];
  float g = bn1[c], bb = bn1[64 + c], mm = bn1[128 + c], vv = bn1[192 + c];
  h[i] = fmaxf(g * (v - mm) * rsqrtf(vv + BN_EPS_) + bb, 0.f);
}

__global__ void k_make16(const float* __restrict__ h, _Float16* __restrict__ cn,
                         _Float16* __restrict__ nc, int C) {
  size_t i = (size_t)blockIdx.x * 256 + threadIdx.x;
  size_t total = (size_t)B_ * C * N_;
  if (i >= total) return;
  int n = (int)(i % N_);
  int c = (int)((i / N_) % C);
  size_t b = i / ((size_t)C * N_);
  _Float16 v = (_Float16)h[i];
  cn[i] = v;
  nc[(b * N_ + n) * C + c] = v;
}

__global__ void k_f2h(const float* __restrict__ src, _Float16* __restrict__ dst, int count) {
  int i = blockIdx.x * 256 + threadIdx.x;
  if (i < count) dst[i] = (_Float16)src[i];
}

// relu(bn(W @ h)) -- 1x4 blocked WMMA; fuse_max folds the 4 point tiles + wave-max
// + single atomicMax per channel (conv5 output never touches HBM).
__global__ void k_conv(const _Float16* __restrict__ w16, const _Float16* __restrict__ h16nc,
                       const float* __restrict__ bnp, float* __restrict__ hout,
                       unsigned* __restrict__ h5max, int Cin, int Cout, int fuse_max) {
  int b = blockIdx.z, c0 = blockIdx.y * 16, p0 = blockIdx.x * 64;
  const _Float16* Bm = h16nc + (size_t)b * N_ * Cin;
  V8FU acc[4];
#pragma unroll
  for (int j = 0; j < 4; ++j) acc[j].v = (v8f){};
  for (int k = 0; k < Cin; k += 32) {
    v16h a = ld_frag(w16, c0, Cin, k);
#pragma unroll
    for (int j = 0; j < 4; ++j) {
      v16h bf = ld_frag(Bm, p0 + j * 16, Cin, k);
      acc[j].v = WMMA16(a, bf, acc[j].v);
    }
  }
  int lane = threadIdx.x;
  int pl = lane & 15;
  int chalf = (lane >> 4) * 8;
  if (fuse_max) {
    float tm[8];
#pragma unroll
    for (int r = 0; r < 8; ++r) tm[r] = 0.f;
#pragma unroll
    for (int j = 0; j < 4; ++j)
#pragma unroll
      for (int r = 0; r < 8; ++r) {
        int c = c0 + r + chalf;
        float g = bnp[c], bb = bnp[Cout + c], mm = bnp[2 * Cout + c], vv = bnp[3 * Cout + c];
        float t = fmaxf(g * (acc[j].f[r] - mm) * rsqrtf(vv + BN_EPS_) + bb, 0.f);
        tm[r] = fmaxf(tm[r], t);
      }
#pragma unroll
    for (int r = 0; r < 8; ++r) {
      float t = tm[r];
      for (int off = 8; off; off >>= 1) t = fmaxf(t, __shfl_xor(t, off));
      if (pl == 0) atomicMax(&h5max[b * E_ + c0 + r + chalf], __float_as_uint(t));
    }
  } else {
#pragma unroll
    for (int j = 0; j < 4; ++j) {
      int p = p0 + j * 16 + pl;
#pragma unroll
      for (int r = 0; r < 8; ++r) {
        int c = c0 + r + chalf;
        float g = bnp[c], bb = bnp[Cout + c], mm = bnp[2 * Cout + c], vv = bnp[3 * Cout + c];
        float t = fmaxf(g * (acc[j].f[r] - mm) * rsqrtf(vv + BN_EPS_) + bb, 0.f);
        hout[((size_t)b * Cout + c) * N_ + p] = t;
      }
    }
  }
}

// ---------------- head ----------------

__global__ void k_head(const unsigned* __restrict__ h5max_u, const float* __restrict__ lin1,
                       const float* __restrict__ bn6, const float* __restrict__ lin2,
                       const float* __restrict__ lin2b, float* __restrict__ out) {
  __shared__ float hb[E_];
  __shared__ float l1[512];
  int b = blockIdx.x, tid = threadIdx.x;
  for (int i = tid; i < E_; i += 256) hb[i] = __uint_as_float(h5max_u[b * E_ + i]);
  __syncthreads();
  for (int o = tid; o < 512; o += 256) {
    float s = 0.f;
    const float* wrow = lin1 + (size_t)o * E_;
    for (int k = 0; k < E_; ++k) s += wrow[k] * hb[k];
    float g = bn6[o], bb = bn6[512 + o], mm = bn6[1024 + o], vv = bn6[1536 + o];
    l1[o] = fmaxf(g * (s - mm) * rsqrtf(vv + BN_EPS_) + bb, 0.f);
  }
  __syncthreads();
  if (tid < 40) {
    float s = 0.f;
    const float* wrow = lin2 + tid * 512;
    for (int k = 0; k < 512; ++k) s += wrow[k] * l1[k];
    out[b * 40 + tid] = s + lin2b[tid];
  }
}

// ---------------- launch ----------------

extern "C" void kernel_launch(void* const* d_in, const int* in_sizes, int n_in,
                              void* d_out, int out_size, void* d_ws, size_t ws_size,
                              hipStream_t stream) {
  (void)in_sizes; (void)n_in; (void)out_size; (void)ws_size;
  const float* x   = (const float*)d_in[0];
  const float* w1  = (const float*)d_in[1];
  const float* convW[4]  = {(const float*)d_in[2], (const float*)d_in[3],
                            (const float*)d_in[4], (const float*)d_in[5]};
  const float* bn1 = (const float*)d_in[6];
  const float* convBN[4] = {(const float*)d_in[7], (const float*)d_in[8],
                            (const float*)d_in[9], (const float*)d_in[10]};
  const float* da_wqk[4] = {(const float*)d_in[11], (const float*)d_in[15],
                            (const float*)d_in[19], (const float*)d_in[23]};
  const float* da_wt[4]  = {(const float*)d_in[12], (const float*)d_in[16],
                            (const float*)d_in[20], (const float*)d_in[24]};
  const float* da_bt[4]  = {(const float*)d_in[13], (const float*)d_in[17],
                            (const float*)d_in[21], (const float*)d_in[25]};
  const float* da_bn[4]  = {(const float*)d_in[14], (const float*)d_in[18],
                            (const float*)d_in[22], (const float*)d_in[26]};
  const float* lin1_w = (const float*)d_in[27];
  const float* bn6    = (const float*)d_in[28];
  const float* lin2_w = (const float*)d_in[29];
  const float* lin2_b = (const float*)d_in[30];
  float* out = (float*)d_out;

  char* ws = (char*)d_ws;
  size_t off = 0;
  auto alloc = [&](size_t bytes) -> void* {
    void* p = ws + off;
    off += (bytes + 255) & ~(size_t)255;
    return p;
  };
  float*    dens   = (float*)alloc((size_t)B_ * N_ * 4);
  float*    nrm    = (float*)alloc((size_t)B_ * N_ * 4);
  float*    rowZ   = (float*)alloc((size_t)B_ * N_ * 4);
  float*    colsum = (float*)alloc((size_t)B_ * N_ * 4);
  float*    dmaxb  = (float*)alloc(256);
  unsigned* gmax   = (unsigned*)alloc(256);
  float*    Sslot  = (float*)alloc(256);
  unsigned* h5max  = (unsigned*)alloc((size_t)B_ * E_ * 4);
  float*    hA     = (float*)alloc((size_t)B_ * CMAX_ * N_ * 4);
  float*    hB     = (float*)alloc((size_t)B_ * CMAX_ * N_ * 4);
  float*    xr     = (float*)alloc((size_t)B_ * CMAX_ * N_ * 4);
  _Float16* h16cn  = (_Float16*)alloc((size_t)B_ * CMAX_ * N_ * 2);
  _Float16* h16nc  = (_Float16*)alloc((size_t)B_ * CMAX_ * N_ * 2);
  _Float16* d16nc  = (_Float16*)alloc((size_t)B_ * CMAX_ * N_ * 2);
  _Float16* wt16   = (_Float16*)alloc((size_t)CMAX_ * CMAX_ * 2);
  _Float16* wcv16  = (_Float16*)alloc((size_t)E_ * CMAX_ * 2);
  _Float16* w16    = (_Float16*)alloc((size_t)B_ * N_ * N_ * 2);  // 128 MB, L2-resident

  dim3 blk256(256);
  dim3 wave(32);

  k_init<<<(B_ * E_ + 255) / 256, blk256, 0, stream>>>(gmax, h5max);
  k_norms<<<(B_ * N_) / 256, blk256, 0, stream>>>(x, nrm);
  k_density<<<B_ * N_, blk256, 0, stream>>>(x, nrm, dens, gmax);
  k_normalize<<<(B_ * N_) / 256, blk256, 0, stream>>>(dens, gmax);
  k_batchmax<<<B_, blk256, 0, stream>>>(dens, dmaxb);

  k_conv1<<<(B_ * 64 * N_) / 256, blk256, 0, stream>>>(x, w1, bn1, hA);
  k_make16<<<(B_ * 64 * N_) / 256, blk256, 0, stream>>>(hA, h16cn, h16nc, 64);

  int attnC[4]    = {64, 64, 64, 128};
  int convCout[4] = {64, 64, 128, 1024};
  int c4s[4]      = {16, 16, 16, 32};

  float* hcur = hA;   // attn input
  float* hnext = hB;  // attn output / conv input

  for (int L = 0; L < 4; ++L) {
    int C = attnC[L];
    size_t tot = (size_t)B_ * C * N_;
    k_sumsq<<<1, 32, 0, stream>>>(da_wqk[L], c4s[L], Sslot);
    k_rowZ<<<(B_ * N_) / 256, blk256, 0, stream>>>(dens, dmaxb, Sslot, rowZ);
    k_build_w<<<B_ * (N_ / 8), blk256, 0, stream>>>(dens, rowZ, dmaxb, Sslot, w16, colsum);
    k_f2h<<<(C * C + 255) / 256, blk256, 0, stream>>>(da_wt[L], wt16, C * C);
    k_xr<<<dim3(N_ / 64, C / 32, B_), wave, 0, stream>>>(h16cn, w16, colsum, xr, C);
    k_diff16<<<(unsigned)((tot + 255) / 256), blk256, 0, stream>>>(hcur, xr, d16nc, C);
    k_post<<<dim3(N_ / 64, C / 16, B_), wave, 0, stream>>>(wt16, d16nc, hcur, da_bt[L], da_bn[L], hnext, C);

    int Cout = convCout[L];
    k_f2h<<<(Cout * C + 255) / 256, blk256, 0, stream>>>(convW[L], wcv16, Cout * C);
    k_make16<<<(unsigned)((tot + 255) / 256), blk256, 0, stream>>>(hnext, h16cn, h16nc, C);
    int fuse = (L == 3) ? 1 : 0;
    k_conv<<<dim3(N_ / 64, Cout / 16, B_), wave, 0, stream>>>(wcv16, h16nc, convBN[L], hcur, h5max, C, Cout, fuse);
    if (!fuse) {
      size_t tot2 = (size_t)B_ * Cout * N_;
      k_make16<<<(unsigned)((tot2 + 255) / 256), blk256, 0, stream>>>(hcur, h16cn, h16nc, Cout);
    }
  }

  k_head<<<B_, blk256, 0, stream>>>(h5max, lin1_w, bn6, lin2_w, lin2_b, out);
}
